// _LoRA_qkv_conv__77756087927069
// MI455X (gfx1250) — compile-verified, matrix operand order checked by reading, not run on hardware
//
#include <hip/hip_runtime.h>
#include <hip/hip_bf16.h>

// ---------------- types ----------------
typedef __attribute__((ext_vector_type(16))) __bf16 bf16x16;
typedef __attribute__((ext_vector_type(8)))  __bf16 bf16x8;
typedef __attribute__((ext_vector_type(4)))  __bf16 bf16x4;
typedef __attribute__((ext_vector_type(8)))  float  f32x8;

// B=8 H=64 W=64 DIM=768 R=16 ; Npix = 32768 ; Kext = 800 ; Nout = 2304
#define NPIX   32768
#define DIM    768
#define NOUT   2304
#define KEXT   800
#define LDSTR  40          // bf16 elements per LDS row (80 bytes, 16B aligned, bank-skewed)

static __device__ inline f32x8 wmma_bf16(bf16x16 a, bf16x16 b, f32x8 c) {
  return __builtin_amdgcn_wmma_f32_16x16x32_bf16(false, a, false, b, (short)0, c, false, false);
}

// Load one 16x32 bf16 fragment (A or B side, identical lane mapping):
// lane<16 -> K chunks {0..7, 16..23}; lane>=16 -> {8..15, 24..31}
static __device__ inline bf16x16 frag16(const __bf16* rowp, int lane) {
  const int c0 = (lane >> 4) << 3;            // 0 or 8
  bf16x8 lo = *(const bf16x8*)(rowp + c0);
  bf16x8 hi = *(const bf16x8*)(rowp + c0 + 16);
  return __builtin_shufflevector(lo, hi, 0,1,2,3,4,5,6,7,8,9,10,11,12,13,14,15);
}

// ---------------- kernel 0a: build K-extended transposed weight Wt[2304][800] bf16 ----
__global__ __launch_bounds__(256) void k_build_wt(const float* __restrict__ Wqkv,
                                                  const float* __restrict__ Wbq,
                                                  const float* __restrict__ Wbv,
                                                  __bf16* __restrict__ Wt) {
  int idx = blockIdx.x * 256 + threadIdx.x;
  if (idx >= NOUT * KEXT) return;
  int n = idx / KEXT, k = idx % KEXT;
  float v;
  if (k < DIM)         v = Wqkv[(size_t)k * NOUT + n];
  else if (k < DIM+16) v = (n < DIM)    ? Wbq[(size_t)(k - DIM) * DIM + n]                : 0.f;
  else                 v = (n >= 2*DIM) ? Wbv[(size_t)(k - DIM - 16) * DIM + (n - 2*DIM)] : 0.f;
  Wt[(size_t)n * KEXT + k] = (__bf16)v;
}

// ---------------- kernel 0b: fold conv kernels/biases; build down-proj weight bf16 ----
__global__ __launch_bounds__(256) void k_build_small(const float* __restrict__ Kq3, const float* __restrict__ bq3,
                                                     const float* __restrict__ Kq1, const float* __restrict__ bq1,
                                                     const float* __restrict__ Kv3, const float* __restrict__ bv3,
                                                     const float* __restrict__ Kv1, const float* __restrict__ bv1,
                                                     const float* __restrict__ Waq, const float* __restrict__ Wav,
                                                     float* __restrict__ keff, float* __restrict__ beff,
                                                     __bf16* __restrict__ Wta) {
  const int t = threadIdx.x;
  // keff[qv][tap][ci][co] ; fold 1x1 conv + identity into center tap (tap==4)
  for (int i = t; i < 2 * 9 * 16 * 16; i += 256) {
    int qv = i / 2304, r = i % 2304;
    int tap = r >> 8, ci = (r >> 4) & 15, co = r & 15;
    const float* K3 = qv ? Kv3 : Kq3;
    const float* K1 = qv ? Kv1 : Kq1;
    float v = K3[r];
    if (tap == 4) v += K1[ci * 16 + co] + ((ci == co) ? 1.f : 0.f);
    keff[i] = v;
  }
  for (int i = t; i < 32; i += 256) {
    int qv = i >> 4, c = i & 15;
    beff[i] = qv ? (bv3[c] + bv1[c]) : (bq3[c] + bq1[c]);
  }
  // Wta[n][k] = (n<16 ? Wa_q : Wa_v)[k][n%16] , n in 0..31, k in 0..767
  for (int i = t; i < 32 * DIM; i += 256) {
    int n = i / DIM, k = i % DIM;
    float v = (n < 16) ? Waq[k * 16 + n] : Wav[k * 16 + (n - 16)];
    Wta[i] = (__bf16)v;
  }
}

// ---------------- kernel 1: a = x @ [Wa_q|Wa_v]  (32768x768x32, WMMA bf16) ------------
// Double-buffered LDS, one barrier per K-step, global prefetch into registers.
__global__ __launch_bounds__(256) void k_down(const float* __restrict__ x,
                                              const __bf16* __restrict__ Wta,
                                              float* __restrict__ a) {
  __shared__ __bf16 sA[2][128 * LDSTR];
  __shared__ __bf16 sB[2][32 * LDSTR];
  const int t = threadIdx.x, lane = t & 31, w = t >> 5;   // 8 waves, each 16 rows x 32 cols
  const int rowBlock = blockIdx.x * 128;
  const int lr = t >> 1, lc0 = (t & 1) * 16;

  const float*  gA = x   + (size_t)(rowBlock + lr) * DIM + lc0;
  const __bf16* gB = Wta + (size_t)lr * DIM + lc0;        // valid only for t < 64

  f32x8 acc0 = {}; f32x8 acc1 = {};
  float4 pa[4]; bf16x8 pb[2];

  auto fetch = [&](int k0) {
    const float4* g = (const float4*)(gA + k0);
    pa[0] = g[0]; pa[1] = g[1]; pa[2] = g[2]; pa[3] = g[3];
    if (t < 64) { const bf16x8* gb = (const bf16x8*)(gB + k0); pb[0] = gb[0]; pb[1] = gb[1]; }
  };
  auto commit = [&](int buf) {
    __bf16* da = &sA[buf][lr * LDSTR + lc0];
    #pragma unroll
    for (int j = 0; j < 4; ++j) {
      float4 f = pa[j];
      bf16x4 hv = {(__bf16)f.x, (__bf16)f.y, (__bf16)f.z, (__bf16)f.w};
      *(bf16x4*)(da + 4 * j) = hv;
    }
    if (t < 64) {
      __bf16* db = &sB[buf][lr * LDSTR + lc0];
      *(bf16x8*)db = pb[0]; *(bf16x8*)(db + 8) = pb[1];
    }
  };

  fetch(0); commit(0);
  __syncthreads();

  for (int ks = 0; ks < DIM / 32; ++ks) {
    const int cur = ks & 1;
    const bool haveNext = (ks + 1) < DIM / 32;
    if (haveNext) fetch((ks + 1) * 32);                  // loads in flight under the WMMAs
    bf16x16 af = frag16(&sA[cur][(w * 16 + (lane & 15)) * LDSTR], lane);
    bf16x16 b0 = frag16(&sB[cur][((lane & 15)) * LDSTR], lane);
    bf16x16 b1 = frag16(&sB[cur][(16 + (lane & 15)) * LDSTR], lane);
    acc0 = wmma_bf16(af, b0, acc0);
    acc1 = wmma_bf16(af, b1, acc1);
    if (haveNext) commit(cur ^ 1);
    __syncthreads();
  }

  const int r0 = rowBlock + w * 16 + ((lane >> 4) << 3);
  const int c = lane & 15;
  #pragma unroll
  for (int r = 0; r < 8; ++r) {
    a[(size_t)(r0 + r) * 32 + c]      = acc0[r];
    a[(size_t)(r0 + r) * 32 + 16 + c] = acc1[r];
  }
}

// ---------------- kernel 2: folded 3x3 conv (+1x1 + identity + biases), bf16 out ------
__global__ __launch_bounds__(256) void k_conv(const float* __restrict__ a,
                                              const float* __restrict__ keff,
                                              const float* __restrict__ beff,
                                              __bf16* __restrict__ mid) {
  __shared__ float sK[2 * 9 * 16 * 16];   // 18 KB
  __shared__ float sT[3 * 18 * 32];       // 6.75 KB: 3 rows x 18 cols x (16 q + 16 v) ch
  const int t = threadIdx.x;
  const int wt = blockIdx.x, h = blockIdx.y, b = blockIdx.z;
  for (int i = t; i < 2 * 9 * 16 * 16; i += 256) sK[i] = keff[i];
  for (int i = t; i < 3 * 18 * 32; i += 256) {
    int ch = i & 31, col = (i >> 5) % 18, rr = (i >> 5) / 18;
    int hh = h - 1 + rr, ww = wt * 16 - 1 + col;
    float v = 0.f;
    if (hh >= 0 && hh < 64 && ww >= 0 && ww < 64)
      v = a[(size_t)(((b * 64 + hh) * 64) + ww) * 32 + ch];
    sT[i] = v;
  }
  __syncthreads();
  const int c = t & 15, px = t >> 4;
  #pragma unroll
  for (int qv = 0; qv < 2; ++qv) {
    float acc = beff[qv * 16 + c];
    #pragma unroll
    for (int ky = 0; ky < 3; ++ky)
      #pragma unroll
      for (int kx = 0; kx < 3; ++kx) {
        const float* ain = &sT[((ky * 18) + (px + kx)) * 32 + qv * 16];
        const float* kk  = &sK[qv * 2304 + ((ky * 3 + kx) * 16) * 16 + c];
        #pragma unroll
        for (int ci = 0; ci < 16; ++ci) acc += ain[ci] * kk[ci * 16];
      }
    int p = ((b * 64 + h) * 64) + wt * 16 + px;
    mid[(size_t)p * 32 + qv * 16 + c] = (__bf16)acc;
  }
}

// ---------------- kernel 3: out = [x|mid] @ Wt' + b_qkv  (32768 x 800 x 2304) ---------
// Double-buffered LDS, one barrier per K-step, NT stores for the 302MB output stream.
__global__ __launch_bounds__(256) void k_gemm(const float* __restrict__ x,
                                              const __bf16* __restrict__ mid,
                                              const __bf16* __restrict__ Wt,
                                              const float* __restrict__ bqkv,
                                              float* __restrict__ out) {
  __shared__ __bf16 sA[2][128 * LDSTR];   // 2 x 10 KB
  __shared__ __bf16 sB[2][128 * LDSTR];   // 2 x 10 KB
  const int t = threadIdx.x, lane = t & 31, w = t >> 5;
  const int wm = w & 3, wn = w >> 2;            // 4x2 wave grid; wave tile 32x64
  const int mBase = blockIdx.x * 128, nBase = blockIdx.y * 128;
  const int lr = t >> 1, lc0 = (t & 1) * 16;

  const float*  gAx = x   + (size_t)(mBase + lr) * DIM  + lc0;
  const __bf16* gAm = mid + (size_t)(mBase + lr) * 32   + lc0;
  const __bf16* gB  = Wt  + (size_t)(nBase + lr) * KEXT + lc0;

  f32x8 acc[8];
  #pragma unroll
  for (int i = 0; i < 8; ++i) acc[i] = (f32x8){};

  float4 pa[4]; bf16x8 pam[2]; bf16x8 pb[2];
  int kPre = 0;

  auto fetch = [&](int k0) {
    kPre = k0;
    if (k0 < DIM) {                              // A from x: fp32 (convert at commit)
      const float4* g = (const float4*)(gAx + k0);
      pa[0] = g[0]; pa[1] = g[1]; pa[2] = g[2]; pa[3] = g[3];
    } else {                                     // A from mid: already bf16
      const bf16x8* g = (const bf16x8*)(gAm + (k0 - DIM));
      pam[0] = g[0]; pam[1] = g[1];
    }
    const bf16x8* gb = (const bf16x8*)(gB + k0);
    pb[0] = gb[0]; pb[1] = gb[1];
  };
  auto commit = [&](int buf) {
    __bf16* da = &sA[buf][lr * LDSTR + lc0];
    if (kPre < DIM) {
      #pragma unroll
      for (int j = 0; j < 4; ++j) {
        float4 f = pa[j];
        bf16x4 hv = {(__bf16)f.x, (__bf16)f.y, (__bf16)f.z, (__bf16)f.w};
        *(bf16x4*)(da + 4 * j) = hv;
      }
    } else {
      *(bf16x8*)da = pam[0]; *(bf16x8*)(da + 8) = pam[1];
    }
    __bf16* db = &sB[buf][lr * LDSTR + lc0];
    *(bf16x8*)db = pb[0]; *(bf16x8*)(db + 8) = pb[1];
  };

  fetch(0); commit(0);
  __syncthreads();

  for (int ks = 0; ks < KEXT / 32; ++ks) {      // 25 K-steps, last one is the LoRA rows
    const int cur = ks & 1;
    const bool haveNext = (ks + 1) < KEXT / 32;
    if (haveNext) fetch((ks + 1) * 32);          // global loads in flight under WMMAs
    bf16x16 af[2], bfr[4];
    #pragma unroll
    for (int mi = 0; mi < 2; ++mi)
      af[mi] = frag16(&sA[cur][(wm * 32 + mi * 16 + (lane & 15)) * LDSTR], lane);
    #pragma unroll
    for (int nj = 0; nj < 4; ++nj)
      bfr[nj] = frag16(&sB[cur][(wn * 64 + nj * 16 + (lane & 15)) * LDSTR], lane);
    #pragma unroll
    for (int mi = 0; mi < 2; ++mi)
      #pragma unroll
      for (int nj = 0; nj < 4; ++nj)
        acc[mi * 4 + nj] = wmma_bf16(af[mi], bfr[nj], acc[mi * 4 + nj]);
    if (haveNext) commit(cur ^ 1);
    __syncthreads();
  }

  // epilogue: + b_qkv; non-temporal stores keep x resident in the 192MB L2
  #pragma unroll
  for (int nj = 0; nj < 4; ++nj) {
    const int col = nBase + wn * 64 + nj * 16 + (lane & 15);
    const float bv = bqkv[col];
    #pragma unroll
    for (int mi = 0; mi < 2; ++mi) {
      const int r0 = mBase + wm * 32 + mi * 16 + ((lane >> 4) << 3);
      f32x8 v = acc[mi * 4 + nj];
      #pragma unroll
      for (int r = 0; r < 8; ++r)
        __builtin_nontemporal_store(v[r] + bv, &out[(size_t)(r0 + r) * NOUT + col]);
    }
  }
}

// ---------------- host launch ----------------
extern "C" void kernel_launch(void* const* d_in, const int* in_sizes, int n_in,
                              void* d_out, int out_size, void* d_ws, size_t ws_size,
                              hipStream_t stream) {
  const float* x    = (const float*)d_in[0];
  const float* Wqkv = (const float*)d_in[1];
  const float* bqkv = (const float*)d_in[2];
  const float* Waq  = (const float*)d_in[3];
  const float* Wbq  = (const float*)d_in[4];
  const float* Wav  = (const float*)d_in[5];
  const float* Wbv  = (const float*)d_in[6];
  const float* Kq3  = (const float*)d_in[7];
  const float* bq3  = (const float*)d_in[8];
  const float* Kv3  = (const float*)d_in[9];
  const float* bv3  = (const float*)d_in[10];
  const float* Kq1  = (const float*)d_in[11];
  const float* bq1  = (const float*)d_in[12];
  const float* Kv1  = (const float*)d_in[13];
  const float* bv1  = (const float*)d_in[14];
  float* out = (float*)d_out;

  char* ws = (char*)d_ws;                 // ~10 MB used, all offsets 16B aligned
  __bf16* Wt   = (__bf16*)(ws + 0);             // 2304*800*2  = 3,686,400
  float*  aLow = (float*) (ws + 3686400);       // 32768*32*4  = 4,194,304
  __bf16* mid  = (__bf16*)(ws + 7880704);       // 32768*32*2  = 2,097,152
  float*  keff = (float*) (ws + 9977856);       // 2*9*16*16*4 = 18,432
  float*  beff = (float*) (ws + 9996288);       // 2*16*4      = 128
  __bf16* Wta  = (__bf16*)(ws + 9996416);       // 32*768*2    = 49,152

  k_build_wt<<<(NOUT * KEXT + 255) / 256, 256, 0, stream>>>(Wqkv, Wbq, Wbv, Wt);
  k_build_small<<<1, 256, 0, stream>>>(Kq3, bq3, Kq1, bq1, Kv3, bv3, Kv1, bv1,
                                       Waq, Wav, keff, beff, Wta);
  k_down<<<NPIX / 128, 256, 0, stream>>>(x, Wta, aLow);
  k_conv<<<dim3(4, 64, 8), 256, 0, stream>>>(aLow, keff, beff, mid);
  k_gemm<<<dim3(NPIX / 128, NOUT / 128), 256, 0, stream>>>(x, mid, Wt, bqkv, out);
}
